// LlamaAttention_23536420782118
// MI455X (gfx1250) — matrix operand order repaired and checked by reading, not compiled, not run on hardware
//
#include <hip/hip_runtime.h>
#include <math.h>

// ---------------------------------------------------------------------------
// Problem constants (from reference): B=1, S=1024, D=4096, 32 heads / 8 KV,
// head_dim=128, ALPHA=5.0, BUDGET=0.2 -> fetch_max=204, NEG=-10000.
// ---------------------------------------------------------------------------
#define SEQ      1024
#define DMODEL   4096
#define NHEADS   32
#define NKV      8
#define HDIM     128
#define FETCHMAX 204
#define ALPHA_C  5.0f
#define SCALING  0.08838834764831845f   // 1/sqrt(128)

typedef unsigned short u16;   // bf16 bit pattern
typedef __attribute__((ext_vector_type(16))) __bf16 v16bf;
typedef __attribute__((ext_vector_type(8)))  float  v8f;

union Frag32B { uint4 u[2]; v16bf v; };

__device__ __forceinline__ u16 f2bf(float f) {
  unsigned int u = __float_as_uint(f);
  unsigned int r = (u + 0x7FFFu + ((u >> 16) & 1u)) >> 16;  // round-nearest-even
  return (u16)r;
}

// ---------------------------------------------------------------------------
// fp32 -> bf16 cast
// ---------------------------------------------------------------------------
__global__ void cast_f32_bf16_kernel(const float* __restrict__ in,
                                     u16* __restrict__ out, int n) {
  int i = blockIdx.x * 256 + threadIdx.x;
  if (i < n) out[i] = f2bf(in[i]);
}

// ---------------------------------------------------------------------------
// Generic bf16 WMMA GEMM:  C(M x N) = alpha * A(M x K) * B(N x K)^T
// M is always 1024 here (gridDim.y == 8, 128 rows/block).
// Block tile 128x64, 8 waves (wave32), each wave 32x32 = 2x2 WMMA 16x16x32.
//
// A tile is staged with GLOBAL_LOAD_ASYNC_TO_LDS_B128 (ASYNCcnt path): each
// lane copies one 16B chunk global->LDS without touching VGPRs, layout
// preserved: As[128][48] (row, k), 96B row stride.
// B tile must be transposed on store (Bs[k][n] = [32][80]), so it uses the
// load->VGPR->ds_store path.
// Next k-tile is prefetched at WGP scope (locality 3) so async copies hit.
//
// Fragment layouts per CDNA5 ISA 7.12.2:
//   A: lane<16 holds row=l, K {0..7,16..23}; lane>=16 row=l-16, K {8..15,24..31}
//   B: lane<16 holds col=l, K {0..7,16..23}; lane>=16 col=l-16, K {8..15,24..31}
//   C: VGPR e -> row e + (lane>=16 ? 8 : 0), col = lane&15
// ---------------------------------------------------------------------------
#define LDS_A_ELEMS (128 * 48)
#define LDS_B_ELEMS (32 * 80)

__global__ __launch_bounds__(256)
void gemm_bf16_wmma(const u16* __restrict__ A, const u16* __restrict__ B,
                    float* __restrict__ C, int K, int lda, int ldb, int ldc,
                    float alpha) {
  // single LDS arena so async-copy byte offsets are well defined (base = 0)
  __shared__ __align__(16) u16 smem[LDS_A_ELEMS + LDS_B_ELEMS];
  u16 (*As)[48] = (u16(*)[48]) & smem[0];
  u16 (*Bs)[80] = (u16(*)[80]) & smem[LDS_A_ELEMS];

  const int tid  = threadIdx.x;
  const int lane = tid & 31;
  const int wave = tid >> 5;
  const int wm   = wave & 3;    // 4 waves along M (32 rows each)
  const int wn   = wave >> 2;   // 2 waves along N (32 cols each)
  const int bm   = blockIdx.y * 128;
  const int bn   = blockIdx.x * 64;
  const int l15  = lane & 15;
  const int hi16 = lane >> 4;   // 0 or 1

  // per-thread staging coordinates (constant across k-loop)
  const int ar0 = (tid) >> 2;             // A chunk 0: row
  const int ac0 = ((tid) & 3) << 3;       //            k element
  const int ar1 = (tid + 256) >> 2;       // A chunk 1
  const int ac1 = ((tid + 256) & 3) << 3;
  const int bn_t = tid >> 2;              // B: n row
  const int bk_t = (tid & 3) << 3;        //    k element

  v8f acc[2][2];
#pragma unroll
  for (int i = 0; i < 2; ++i)
#pragma unroll
    for (int j = 0; j < 2; ++j)
#pragma unroll
      for (int e = 0; e < 8; ++e) acc[i][j][e] = 0.0f;

  for (int k0 = 0; k0 < K; k0 += 32) {
    // ---- prefetch next k-tile into the WGP-near caches while this one runs
    if (k0 + 32 < K) {
      __builtin_prefetch(A + (size_t)(bm + ar0) * lda + (k0 + 32 + ac0), 0, 3);
      __builtin_prefetch(B + (size_t)(bn + bn_t) * ldb + (k0 + 32 + bk_t), 0, 3);
    }

    // ---- stage A tile via async global->LDS copy (no VGPR staging)
    {
      unsigned lds0 = (unsigned)(ar0 * 96 + ac0 * 2);  // byte offset in LDS
      unsigned long long ga0 =
          (unsigned long long)(A + (size_t)(bm + ar0) * lda + (k0 + ac0));
      asm volatile("global_load_async_to_lds_b128 %0, %1, off"
                   :: "v"(lds0), "v"(ga0) : "memory");
      unsigned lds1 = (unsigned)(ar1 * 96 + ac1 * 2);
      unsigned long long ga1 =
          (unsigned long long)(A + (size_t)(bm + ar1) * lda + (k0 + ac1));
      asm volatile("global_load_async_to_lds_b128 %0, %1, off"
                   :: "v"(lds1), "v"(ga1) : "memory");
    }

    // ---- stage B tile transposed: Bs[k][n] = B[bn+n][k0+k]
    {
      uint4 raw = *(const uint4*)(B + (size_t)(bn + bn_t) * ldb + (k0 + bk_t));
      const u16* bv = (const u16*)&raw;
#pragma unroll
      for (int j = 0; j < 8; ++j) Bs[bk_t + j][bn_t] = bv[j];
    }

    asm volatile("s_wait_asynccnt 0x0" ::: "memory");
    __syncthreads();

    Frag32B fa[2], fb[2];
#pragma unroll
    for (int t = 0; t < 2; ++t) {
      int r  = 32 * wm + 16 * t + l15;
      int kb = hi16 << 3;                      // 0 or 8
      fa[t].u[0] = *(const uint4*)&As[r][kb];
      fa[t].u[1] = *(const uint4*)&As[r][kb + 16];
      int n0 = 32 * wn + 16 * t;
#pragma unroll
      for (int s = 0; s < 16; ++s)
        ((u16*)&fb[t])[s] = Bs[kb + (s & 7) + ((s >> 3) << 4)][n0 + l15];
    }
#pragma unroll
    for (int i = 0; i < 2; ++i)
#pragma unroll
      for (int j = 0; j < 2; ++j)
        acc[i][j] = __builtin_amdgcn_wmma_f32_16x16x32_bf16(
            false, fa[i].v, false, fb[j].v, (short)0, acc[i][j], false, false);
    __syncthreads();
  }

#pragma unroll
  for (int i = 0; i < 2; ++i)
#pragma unroll
    for (int j = 0; j < 2; ++j)
#pragma unroll
      for (int e = 0; e < 8; ++e) {
        int row = bm + 32 * wm + 16 * i + e + hi16 * 8;
        int col = bn + 32 * wn + 16 * j + l15;
        C[(size_t)row * ldc + col] = alpha * acc[i][j][e];
      }
}

// ---------------------------------------------------------------------------
// RoPE + repack to head-major bf16: out[h][s][d] for d in [0,64) pairs
// X is (s, n_heads*128) fp32.
// ---------------------------------------------------------------------------
__global__ void rope_repack_kernel(const float* __restrict__ X,
                                   const float* __restrict__ cosp,
                                   const float* __restrict__ sinp,
                                   u16* __restrict__ out, int n_heads) {
  int idx   = blockIdx.x * 256 + threadIdx.x;
  int total = SEQ * n_heads * 64;
  if (idx >= total) return;
  int d = idx & 63;
  int t = idx >> 6;
  int h = t % n_heads;
  int s = t / n_heads;
  int stride = n_heads * HDIM;
  float x1 = X[(size_t)s * stride + h * HDIM + d];
  float x2 = X[(size_t)s * stride + h * HDIM + d + 64];
  float c1 = cosp[s * HDIM + d],      s1 = sinp[s * HDIM + d];
  float c2 = cosp[s * HDIM + d + 64], s2 = sinp[s * HDIM + d + 64];
  float o1 = x1 * c1 - x2 * s1;   // rotate_half: first half gets -x2
  float o2 = x2 * c2 + x1 * s2;
  size_t ob = ((size_t)h * SEQ + s) * HDIM;
  out[ob + d]      = f2bf(o1);
  out[ob + d + 64] = f2bf(o2);
}

// V (s, kv*128+d) fp32 -> Vt[kv][d][s] bf16  (transposed for A*B^T GEMM form)
__global__ void v_repack_kernel(const float* __restrict__ V,
                                u16* __restrict__ Vt) {
  int idx = blockIdx.x * 256 + threadIdx.x;
  if (idx >= NKV * HDIM * SEQ) return;
  int s  = idx & (SEQ - 1);
  int t  = idx >> 10;
  int d  = t & (HDIM - 1);
  int kv = t >> 7;
  Vt[idx] = f2bf(V[(size_t)s * (NKV * HDIM) + kv * HDIM + d]);
}

// ---------------------------------------------------------------------------
// Row max over causal region (head 0 scores) -> rowmax0[row]
// ---------------------------------------------------------------------------
__global__ __launch_bounds__(256)
void row_max_kernel(const float* __restrict__ S, float* __restrict__ rowmax) {
  int row = blockIdx.x, tid = threadIdx.x;
  __shared__ float red[256];
  float m = -3.0e38f;
  for (int c = tid; c <= row; c += 256) m = fmaxf(m, S[(size_t)row * SEQ + c]);
  red[tid] = m;
  __syncthreads();
  for (int s = 128; s > 0; s >>= 1) {
    if (tid < s) red[tid] = fmaxf(red[tid], red[tid + s]);
    __syncthreads();
  }
  if (tid == 0) rowmax[row] = red[0];
}

// count per row: #{c<=row : S[row][c] >= rowmax0[row]-ALPHA}
__global__ __launch_bounds__(256)
void count_kernel(const float* __restrict__ S, const float* __restrict__ rowmax0,
                  int* __restrict__ cnt_out) {
  int row = blockIdx.x, tid = threadIdx.x;
  __shared__ int red[256];
  float thr = rowmax0[row] - ALPHA_C;
  int lc = 0;
  for (int c = tid; c <= row; c += 256)
    lc += (S[(size_t)row * SEQ + c] >= thr) ? 1 : 0;
  red[tid] = lc;
  __syncthreads();
  for (int s = 128; s > 0; s >>= 1) {
    if (tid < s) red[tid] += red[tid + s];
    __syncthreads();
  }
  if (tid == 0) cnt_out[row] = red[0];
}

// fetch_num[row] = min(trunc(mean_h counts), 204), clamped >= 1
__global__ void fetch_kernel(const int* __restrict__ counts,
                             int* __restrict__ fetchn) {
  int row = blockIdx.x * 256 + threadIdx.x;
  if (row >= SEQ) return;
  float s = 0.0f;
  for (int h = 0; h < NHEADS; ++h) s += (float)counts[h * SEQ + row];
  int fn = (int)(s * (1.0f / NHEADS));
  if (fn > FETCHMAX) fn = FETCHMAX;
  if (fn < 1) fn = 1;
  fetchn[row] = fn;
}

// ---------------------------------------------------------------------------
// Per-row top-k mask + softmax, writes probs as bf16.
// Top-k threshold found by 24-step bisection on value (approximates
// ranks < fetch_num from the reference; rows < fetch_max keep full causal).
// ---------------------------------------------------------------------------
__global__ __launch_bounds__(256)
void masked_softmax_kernel(const float* __restrict__ S,
                           const int* __restrict__ fetchn,
                           u16* __restrict__ P) {
  int row = blockIdx.x, tid = threadIdx.x;
  const float* Sr = S + (size_t)row * SEQ;
  u16* Pr = P + (size_t)row * SEQ;
  int ncol = row + 1;
  __shared__ float redf[256];
  __shared__ int   redi[256];

  float lmax = -3.0e38f, lmin = 3.0e38f;
  for (int c = tid; c < ncol; c += 256) {
    float v = Sr[c];
    lmax = fmaxf(lmax, v);
    lmin = fminf(lmin, v);
  }
  redf[tid] = lmax;
  __syncthreads();
  for (int s = 128; s > 0; s >>= 1) {
    if (tid < s) redf[tid] = fmaxf(redf[tid], redf[tid + s]);
    __syncthreads();
  }
  float vmax = redf[0];
  __syncthreads();
  redf[tid] = lmin;
  __syncthreads();
  for (int s = 128; s > 0; s >>= 1) {
    if (tid < s) redf[tid] = fminf(redf[tid], redf[tid + s]);
    __syncthreads();
  }
  float vmin = redf[0];
  __syncthreads();

  int fn = fetchn[row];
  float thr = vmin;  // keep everything by default (causal rows)
  if (row >= FETCHMAX && fn < ncol) {
    float lo = vmin, hi = vmax;
    for (int it = 0; it < 24; ++it) {
      float mid = 0.5f * (lo + hi);
      int lc = 0;
      for (int c = tid; c < ncol; c += 256) lc += (Sr[c] >= mid) ? 1 : 0;
      redi[tid] = lc;
      __syncthreads();
      for (int s = 128; s > 0; s >>= 1) {
        if (tid < s) redi[tid] += redi[tid + s];
        __syncthreads();
      }
      int cnt = redi[0];
      __syncthreads();
      if (cnt >= fn) lo = mid; else hi = mid;
    }
    thr = lo;
  }

  float lsum = 0.0f;
  for (int c = tid; c < ncol; c += 256) {
    float v = Sr[c];
    if (v >= thr) lsum += __expf(v - vmax);
  }
  redf[tid] = lsum;
  __syncthreads();
  for (int s = 128; s > 0; s >>= 1) {
    if (tid < s) redf[tid] += redf[tid + s];
    __syncthreads();
  }
  float inv = 1.0f / redf[0];
  __syncthreads();

  for (int c = tid; c < SEQ; c += 256) {
    float p = 0.0f;
    if (c < ncol) {
      float v = Sr[c];
      if (v >= thr) p = __expf(v - vmax) * inv;
    }
    Pr[c] = f2bf(p);
  }
}

// ---------------------------------------------------------------------------
// Host-side orchestration
// ---------------------------------------------------------------------------
extern "C" void kernel_launch(void* const* d_in, const int* in_sizes, int n_in,
                              void* d_out, int out_size, void* d_ws,
                              size_t ws_size, hipStream_t stream) {
  const float* hidden = (const float*)d_in[0];
  const float* cosp   = (const float*)d_in[1];
  const float* sinp   = (const float*)d_in[2];
  const float* Wq     = (const float*)d_in[3];
  const float* Wk     = (const float*)d_in[4];
  const float* Wv     = (const float*)d_in[5];
  const float* Wo     = (const float*)d_in[6];
  float* out = (float*)d_out;

  char* w = (char*)d_ws;
  size_t off = 0;
  auto alloc = [&](size_t bytes) -> void* {
    void* p = (void*)(w + off);
    off += (bytes + 255) & ~(size_t)255;
    return p;
  };

  u16*   hs_bf   = (u16*)alloc((size_t)SEQ * DMODEL * 2);
  u16*   Wq_bf   = (u16*)alloc((size_t)DMODEL * DMODEL * 2);
  u16*   Wk_bf   = (u16*)alloc((size_t)NKV * HDIM * DMODEL * 2);
  u16*   Wv_bf   = (u16*)alloc((size_t)NKV * HDIM * DMODEL * 2);
  u16*   Wo_bf   = (u16*)alloc((size_t)DMODEL * DMODEL * 2);
  float* Qf      = (float*)alloc((size_t)SEQ * DMODEL * 4);
  float* Kf      = (float*)alloc((size_t)SEQ * NKV * HDIM * 4);
  float* Vf      = (float*)alloc((size_t)SEQ * NKV * HDIM * 4);
  u16*   Qbf     = (u16*)alloc((size_t)NHEADS * SEQ * HDIM * 2);
  u16*   Kbf     = (u16*)alloc((size_t)NKV * SEQ * HDIM * 2);
  u16*   Vtbf    = (u16*)alloc((size_t)NKV * HDIM * SEQ * 2);
  float* scores  = (float*)alloc((size_t)SEQ * SEQ * 4);
  u16*   probsbf = (u16*)alloc((size_t)SEQ * SEQ * 2);
  float* attnf   = (float*)alloc((size_t)SEQ * DMODEL * 4);
  u16*   attnbf  = (u16*)alloc((size_t)SEQ * DMODEL * 2);
  float* rowmax0 = (float*)alloc((size_t)SEQ * 4);
  int*   counts  = (int*)alloc((size_t)NHEADS * SEQ * 4);
  int*   fetchn  = (int*)alloc((size_t)SEQ * 4);

  auto cast = [&](const float* in, u16* ot, int n) {
    cast_f32_bf16_kernel<<<(n + 255) / 256, 256, 0, stream>>>(in, ot, n);
  };
  auto gemm = [&](const u16* A, const u16* B, float* C, int N, int K, int lda,
                  int ldb, int ldc, float alpha) {
    dim3 grid(N / 64, SEQ / 128);
    gemm_bf16_wmma<<<grid, 256, 0, stream>>>(A, B, C, K, lda, ldb, ldc, alpha);
  };

  // --- bf16 conversions -----------------------------------------------------
  cast(hidden, hs_bf, SEQ * DMODEL);
  cast(Wq, Wq_bf, DMODEL * DMODEL);
  cast(Wk, Wk_bf, NKV * HDIM * DMODEL);
  cast(Wv, Wv_bf, NKV * HDIM * DMODEL);
  cast(Wo, Wo_bf, DMODEL * DMODEL);

  // --- QKV projections ------------------------------------------------------
  gemm(hs_bf, Wq_bf, Qf, DMODEL, DMODEL, DMODEL, DMODEL, DMODEL, 1.0f);
  gemm(hs_bf, Wk_bf, Kf, NKV * HDIM, DMODEL, DMODEL, DMODEL, NKV * HDIM, 1.0f);
  gemm(hs_bf, Wv_bf, Vf, NKV * HDIM, DMODEL, DMODEL, DMODEL, NKV * HDIM, 1.0f);

  // --- RoPE + repacking -----------------------------------------------------
  rope_repack_kernel<<<(SEQ * NHEADS * 64 + 255) / 256, 256, 0, stream>>>(
      Qf, cosp, sinp, Qbf, NHEADS);
  rope_repack_kernel<<<(SEQ * NKV * 64 + 255) / 256, 256, 0, stream>>>(
      Kf, cosp, sinp, Kbf, NKV);
  v_repack_kernel<<<(NKV * HDIM * SEQ + 255) / 256, 256, 0, stream>>>(Vf, Vtbf);

  // --- Phase 1: head-0 scores -> row max ------------------------------------
  gemm(Qbf, Kbf, scores, SEQ, HDIM, HDIM, HDIM, SEQ, SCALING);
  row_max_kernel<<<SEQ, 256, 0, stream>>>(scores, rowmax0);

  // --- Phase 2: per-head counts ---------------------------------------------
  for (int h = 0; h < NHEADS; ++h) {
    const u16* Qh = Qbf + (size_t)h * SEQ * HDIM;
    const u16* Kh = Kbf + (size_t)(h / 4) * SEQ * HDIM;
    gemm(Qh, Kh, scores, SEQ, HDIM, HDIM, HDIM, SEQ, SCALING);
    count_kernel<<<SEQ, 256, 0, stream>>>(scores, rowmax0, counts + h * SEQ);
  }
  fetch_kernel<<<SEQ / 256, 256, 0, stream>>>(counts, fetchn);

  // --- Phase 3: masked softmax + PV per head --------------------------------
  for (int h = 0; h < NHEADS; ++h) {
    const u16* Qh = Qbf + (size_t)h * SEQ * HDIM;
    const u16* Kh = Kbf + (size_t)(h / 4) * SEQ * HDIM;
    const u16* Vh = Vtbf + (size_t)(h / 4) * HDIM * SEQ;
    gemm(Qh, Kh, scores, SEQ, HDIM, HDIM, HDIM, SEQ, SCALING);
    masked_softmax_kernel<<<SEQ, 256, 0, stream>>>(scores, fetchn, probsbf);
    gemm(probsbf, Vh, attnf + (size_t)h * HDIM, HDIM, SEQ, SEQ, SEQ, DMODEL,
         1.0f);
  }

  // --- Output projection ----------------------------------------------------
  cast(attnf, attnbf, SEQ * DMODEL);
  gemm(attnbf, Wo_bf, out, DMODEL, DMODEL, DMODEL, DMODEL, DMODEL, 1.0f);
}
